// SGPool_35811437314383
// MI455X (gfx1250) — matrix operand order repaired and checked by reading, not compiled, not run on hardware
//
#include <hip/hip_runtime.h>

#define B_ 8
#define N_ 4096
#define C_ 64
#define S_ 1024
#define K_ 32
#define M_ (B_*S_)            // 8192 samples
#define SLAB_ (C_*K_)         // 2048 elems per sample
#define NMK_ ((float)(M_*K_)) // BN normalizer

typedef __attribute__((ext_vector_type(16))) __bf16 v16bf;
typedef __attribute__((ext_vector_type(8)))  __bf16 v8bf;
typedef __attribute__((ext_vector_type(8)))  float  v8f;

__device__ __forceinline__ unsigned short f2bf(float f) {
  unsigned u = __builtin_bit_cast(unsigned, f);
  unsigned r = (u + 0x7FFFu + ((u >> 16) & 1u)) >> 16;   // round-to-nearest-even
  return (unsigned short)r;
}

// ---------------------------------------------------------------------------
// 1) Farthest point sampling: one block per batch, sequential 1024 steps.
//    Writes fps order, and new_xyz directly into d_out in [B,3,S] layout.
// ---------------------------------------------------------------------------
__global__ __launch_bounds__(256)
void fps_kernel(const float* __restrict__ xyz, int* __restrict__ fps_idx,
                float* __restrict__ out_xyz /* d_out, [B,3,S] */) {
  const int b = blockIdx.x;
  const int t = threadIdx.x;
  const float* X = xyz + (size_t)b * N_ * 3;

  float px[16], py[16], pz[16], dd[16];
#pragma unroll
  for (int j = 0; j < 16; ++j) {
    int n = j * 256 + t;
    px[j] = X[n * 3 + 0]; py[j] = X[n * 3 + 1]; pz[j] = X[n * 3 + 2];
    dd[j] = 1e10f;
  }

  __shared__ float rd[256];
  __shared__ int   ri[256];
  __shared__ float c3[3];

  int far = 0;
  for (int it = 0; it < S_; ++it) {
    if (t == 0) {
      float cx = X[far * 3 + 0], cy = X[far * 3 + 1], cz = X[far * 3 + 2];
      c3[0] = cx; c3[1] = cy; c3[2] = cz;
      fps_idx[b * S_ + it] = far;
      out_xyz[b * 3 * S_ + 0 * S_ + it] = cx;
      out_xyz[b * 3 * S_ + 1 * S_ + it] = cy;
      out_xyz[b * 3 * S_ + 2 * S_ + it] = cz;
    }
    __syncthreads();
    float cx = c3[0], cy = c3[1], cz = c3[2];
    float best = -1.0f; int bi = 0;
#pragma unroll
    for (int j = 0; j < 16; ++j) {
      float dx = px[j] - cx, dy = py[j] - cy, dz = pz[j] - cz;
      float d = dx * dx + dy * dy + dz * dz;
      dd[j] = fminf(dd[j], d);
      if (dd[j] > best) { best = dd[j]; bi = j * 256 + t; }
    }
    rd[t] = best; ri[t] = bi;
    __syncthreads();
    for (int s = 128; s > 0; s >>= 1) {
      if (t < s && rd[t + s] > rd[t]) { rd[t] = rd[t + s]; ri[t] = ri[t + s]; }
      __syncthreads();
    }
    far = ri[0];
    __syncthreads();
  }
}

// ---------------------------------------------------------------------------
// 2) kNN (32 smallest sq-dists): one block per query; LDS distance array +
//    32 rounds of tree argmin. Neighbor ORDER is irrelevant downstream
//    (conv is per-k, BN over k, final max over k), so any 32-smallest set works.
// ---------------------------------------------------------------------------
__global__ __launch_bounds__(256)
void knn_kernel(const float* __restrict__ xyz, const float* __restrict__ new_xyz,
                int* __restrict__ idx) {
  const int blk = blockIdx.x;          // b*S + s
  const int b = blk >> 10, s = blk & 1023;
  const int t = threadIdx.x;
  const float* X = xyz + (size_t)b * N_ * 3;
  float qx = new_xyz[b * 3 * S_ + 0 * S_ + s];
  float qy = new_xyz[b * 3 * S_ + 1 * S_ + s];
  float qz = new_xyz[b * 3 * S_ + 2 * S_ + s];

  __shared__ float dist[N_];
  __shared__ float rd[256];
  __shared__ int   ri[256];

#pragma unroll
  for (int j = 0; j < 16; ++j) {
    int n = j * 256 + t;
    float dx = X[n * 3 + 0] - qx, dy = X[n * 3 + 1] - qy, dz = X[n * 3 + 2] - qz;
    dist[n] = dx * dx + dy * dy + dz * dz;
  }
  __syncthreads();

  for (int k = 0; k < K_; ++k) {
    float best = 1e30f; int bi = 0;
#pragma unroll
    for (int j = 0; j < 16; ++j) {
      int n = j * 256 + t;
      float d = dist[n];
      if (d < best) { best = d; bi = n; }
    }
    rd[t] = best; ri[t] = bi;
    __syncthreads();
    for (int sr = 128; sr > 0; sr >>= 1) {
      if (t < sr && rd[t + sr] < rd[t]) { rd[t] = rd[t + sr]; ri[t] = ri[t + sr]; }
      __syncthreads();
    }
    if (t == 0) { idx[(size_t)blk * K_ + k] = ri[0]; dist[ri[0]] = 1e30f; }
    __syncthreads();
  }
}

// ---------------------------------------------------------------------------
// 3) Gather features -> bf16 activation buffer A in [m][k][c] layout
//    (c contiguous so WMMA B-fragments are two aligned 16B loads).
// ---------------------------------------------------------------------------
__global__ __launch_bounds__(256)
void gather_kernel(const float* __restrict__ features, const int* __restrict__ idx,
                   unsigned short* __restrict__ A) {
  const size_t total = (size_t)M_ * K_ * C_;
  for (size_t i = (size_t)blockIdx.x * blockDim.x + threadIdx.x; i < total;
       i += (size_t)gridDim.x * blockDim.x) {
    int c = (int)(i & 63);
    size_t mk = i >> 6;               // m*K + k
    size_t m = mk >> 5;
    int b = (int)(m >> 10);
    int n = idx[mk];
    A[i] = f2bf(features[(size_t)b * N_ * C_ + (size_t)n * C_ + c]);
  }
}

// ---------------------------------------------------------------------------
// 4) Convert the 5 weight matrices [o][c] fp32 -> bf16 once.
// ---------------------------------------------------------------------------
__global__ void wconv_kernel(const float* __restrict__ w0, const float* __restrict__ w1,
                             const float* __restrict__ w2, const float* __restrict__ w3,
                             const float* __restrict__ w4, unsigned short* __restrict__ Wb) {
  int i = blockIdx.x * blockDim.x + threadIdx.x;
  if (i < C_ * C_) {
    Wb[0 * 4096 + i] = f2bf(w0[i]);
    Wb[1 * 4096 + i] = f2bf(w1[i]);
    Wb[2 * 4096 + i] = f2bf(w2[i]);
    Wb[3 * 4096 + i] = f2bf(w3[i]);
    Wb[4 * 4096 + i] = f2bf(w4[i]);
  }
}

// ---------------------------------------------------------------------------
// 5) Batched 64x64 @ 64x32 GEMM via v_wmma_f32_16x16x32_bf16.
//    A-operand = W tiles (loaded once per wave, reused across samples),
//    B-operand = activations [m][k][c]. D tiles: 4 o-tiles x 2 k-tiles,
//    each accumulating over 2 bf16 K=32 chunks => 16 WMMA / sample.
// ---------------------------------------------------------------------------
__global__ __launch_bounds__(256)
void conv_wmma_kernel(const unsigned short* __restrict__ Au,
                      const unsigned short* __restrict__ Wu,
                      float* __restrict__ Y /* [m][o][k] */) {
  const __bf16* A = reinterpret_cast<const __bf16*>(Au);
  const __bf16* W = reinterpret_cast<const __bf16*>(Wu);
  const int lane = threadIdx.x & 31;
  const int ll = lane & 15;
  const int hi = lane >> 4;
  const int wave = blockIdx.x * (blockDim.x >> 5) + (threadIdx.x >> 5);
  const int nwaves = gridDim.x * (blockDim.x >> 5);

  // W fragments (16-bit A layout: lanes 0-15 -> K 0..7 & 16..23; lanes 16-31 -> 8..15 & 24..31)
  v16bf wf[4][2];
#pragma unroll
  for (int ob = 0; ob < 4; ++ob) {
    int row = ob * 16 + ll;
#pragma unroll
    for (int cb = 0; cb < 2; ++cb) {
      int c0 = cb * 32 + hi * 8;
      v8bf lo = *(const v8bf*)(W + row * 64 + c0);
      v8bf hi8 = *(const v8bf*)(W + row * 64 + c0 + 16);
      v16bf f;
#pragma unroll
      for (int j = 0; j < 8; ++j) { f[j] = lo[j]; f[8 + j] = hi8[j]; }
      wf[ob][cb] = f;
    }
  }

  for (int m = wave; m < M_; m += nwaves) {
    const __bf16* X = A + (size_t)m * SLAB_;   // [k][c]
    // B fragments: lane n<16 -> col k=kb*16+n, rows c 0..7 & 16..23; lanes 16-31 -> 8..15 & 24..31
    v16bf bfr[2][2];
#pragma unroll
    for (int kb = 0; kb < 2; ++kb) {
      int k = kb * 16 + ll;
#pragma unroll
      for (int cb = 0; cb < 2; ++cb) {
        int c0 = cb * 32 + hi * 8;
        v8bf lo = *(const v8bf*)(X + k * 64 + c0);
        v8bf h8 = *(const v8bf*)(X + k * 64 + c0 + 16);
        v16bf f;
#pragma unroll
        for (int j = 0; j < 8; ++j) { f[j] = lo[j]; f[8 + j] = h8[j]; }
        bfr[kb][cb] = f;
      }
    }
    v8f acc[4][2];
#pragma unroll
    for (int ob = 0; ob < 4; ++ob)
#pragma unroll
      for (int kb = 0; kb < 2; ++kb) {
        v8f c = {};
        c = __builtin_amdgcn_wmma_f32_16x16x32_bf16(false, wf[ob][0], false, bfr[kb][0],
                                                    (short)0, c, false, false);
        c = __builtin_amdgcn_wmma_f32_16x16x32_bf16(false, wf[ob][1], false, bfr[kb][1],
                                                    (short)0, c, false, false);
        acc[ob][kb] = c;
      }
    float* Ym = Y + (size_t)m * SLAB_;
#pragma unroll
    for (int ob = 0; ob < 4; ++ob)
#pragma unroll
      for (int kb = 0; kb < 2; ++kb) {
        v8f c = acc[ob][kb];
        int o0 = ob * 16 + hi * 8;
        int k = kb * 16 + ll;
#pragma unroll
        for (int r = 0; r < 8; ++r)
          Ym[(o0 + r) * K_ + k] = c[r];
      }
  }
}

// ---------------------------------------------------------------------------
// 6) Per-channel sum / sum-of-squares for training-mode BN.
//    Each thread privately reduces one 32-elem run (constant channel),
//    then LDS atomics + one global atomic per channel per block.
// ---------------------------------------------------------------------------
__global__ __launch_bounds__(256)
void stats_kernel(const float* __restrict__ Y, float* __restrict__ sum, float* __restrict__ sq) {
  __shared__ float ssum[C_], ssq[C_];
  if (threadIdx.x < C_) { ssum[threadIdx.x] = 0.f; ssq[threadIdx.x] = 0.f; }
  __syncthreads();
  const int nruns = M_ * C_;
  for (int run = blockIdx.x * blockDim.x + threadIdx.x; run < nruns;
       run += gridDim.x * blockDim.x) {
    const float* p = Y + (size_t)run * K_;
    float s = 0.f, q = 0.f;
#pragma unroll
    for (int k = 0; k < K_; ++k) { float v = p[k]; s += v; q += v * v; }
    int c = run & 63;
    atomicAdd(&ssum[c], s);
    atomicAdd(&ssq[c], q);
  }
  __syncthreads();
  if (threadIdx.x < C_) {
    atomicAdd(&sum[threadIdx.x], ssum[threadIdx.x]);
    atomicAdd(&sq[threadIdx.x], ssq[threadIdx.x]);
  }
}

// 7) Fold mean/var/gamma/beta into per-channel scale/shift. (Conv bias cancels
//    exactly under training-mode BN, so it is omitted.)
__global__ void bnscale_kernel(const float* __restrict__ sum, const float* __restrict__ sq,
                               const float* __restrict__ g, const float* __restrict__ beta,
                               float* __restrict__ scale, float* __restrict__ shift) {
  int c = threadIdx.x;
  if (c < C_) {
    float mean = sum[c] / NMK_;
    float var = sq[c] / NMK_ - mean * mean;
    float inv = rsqrtf(var + 1e-5f);
    float sc = g[c] * inv;
    scale[c] = sc;
    shift[c] = beta[c] - mean * sc;
  }
}

// ---------------------------------------------------------------------------
// 8) BN apply (+optional residual) + leakyReLU; writes optional fp32 copy
//    (residual chain / final) and the transposed bf16 input for next GEMM
//    via an LDS transpose tile. One block per sample m.
// ---------------------------------------------------------------------------
__global__ __launch_bounds__(256)
void bnapply_kernel(const float* __restrict__ Y, const float* __restrict__ scale,
                    const float* __restrict__ shift, const float* __restrict__ res,
                    float* __restrict__ outF, unsigned short* __restrict__ outA) {
  __shared__ float tile[SLAB_];
  const int m = blockIdx.x;
  const float* Ym = Y + (size_t)m * SLAB_;
  const float* Rm = res ? res + (size_t)m * SLAB_ : nullptr;
  for (int i = threadIdx.x; i < SLAB_; i += 256) {
    int o = i >> 5;
    float v = scale[o] * Ym[i] + shift[o];
    if (Rm) v += Rm[i];
    v = (v >= 0.f) ? v : 0.1f * v;
    tile[i] = v;
  }
  __syncthreads();
  float* Om = outF ? outF + (size_t)m * SLAB_ : nullptr;
  unsigned short* Am = outA + (size_t)m * SLAB_;
  for (int i = threadIdx.x; i < SLAB_; i += 256) {
    if (Om) Om[i] = tile[i];
    int k = i >> 6, c = i & 63;               // out layout [k][c]
    Am[i] = f2bf(tile[c * K_ + k]);
  }
}

// 9) Max over K per (m, c) -> d_out [B, C, S]
__global__ __launch_bounds__(256)
void maxpool_kernel(const float* __restrict__ Y, float* __restrict__ out) {
  const int total = M_ * C_;
  for (int i = blockIdx.x * blockDim.x + threadIdx.x; i < total;
       i += gridDim.x * blockDim.x) {
    int c = i & 63, m = i >> 6;
    const float* p = Y + (size_t)m * SLAB_ + c * K_;
    float v = p[0];
#pragma unroll
    for (int k = 1; k < K_; ++k) v = fmaxf(v, p[k]);
    int b = m >> 10, s = m & 1023;
    out[B_ * 3 * S_ + (size_t)b * C_ * S_ + (size_t)c * S_ + s] = v;
  }
}

// ---------------------------------------------------------------------------
extern "C" void kernel_launch(void* const* d_in, const int* in_sizes, int n_in,
                              void* d_out, int out_size, void* d_ws, size_t ws_size,
                              hipStream_t stream) {
  (void)in_sizes; (void)n_in; (void)out_size; (void)ws_size;
  const float* xyz      = (const float*)d_in[0];
  const float* features = (const float*)d_in[1];
  const float* w[5]  = {(const float*)d_in[2],  (const float*)d_in[3],  (const float*)d_in[4],
                        (const float*)d_in[5],  (const float*)d_in[6]};
  const float* g[5]  = {(const float*)d_in[12], (const float*)d_in[13], (const float*)d_in[14],
                        (const float*)d_in[15], (const float*)d_in[16]};
  const float* be[5] = {(const float*)d_in[17], (const float*)d_in[18], (const float*)d_in[19],
                        (const float*)d_in[20], (const float*)d_in[21]};
  float* out = (float*)d_out;
  char* ws = (char*)d_ws;

  size_t off = 0;
  auto carve = [&](size_t bytes) -> char* {
    char* p = ws + off;
    off += (bytes + 255) & ~(size_t)255;
    return p;
  };
  unsigned short* Wb   = (unsigned short*)carve(5 * 4096 * sizeof(unsigned short));
  int*            fidx = (int*)carve((size_t)B_ * S_ * sizeof(int));
  int*            nidx = (int*)carve((size_t)M_ * K_ * sizeof(int));
  float*          stat = (float*)carve(5 * 256 * sizeof(float));   // per layer: sum|sq|scale|shift
  unsigned short* A    = (unsigned short*)carve((size_t)M_ * SLAB_ * sizeof(unsigned short));
  float*          Y    = (float*)carve((size_t)M_ * SLAB_ * sizeof(float));
  float*          R    = (float*)carve((size_t)M_ * SLAB_ * sizeof(float));

  hipMemsetAsync(stat, 0, 5 * 256 * sizeof(float), stream);

  wconv_kernel<<<16, 256, 0, stream>>>(w[0], w[1], w[2], w[3], w[4], Wb);
  fps_kernel<<<B_, 256, 0, stream>>>(xyz, fidx, out);
  knn_kernel<<<M_, 256, 0, stream>>>(xyz, out, nidx);
  gather_kernel<<<8192, 256, 0, stream>>>(features, nidx, A);

  // Per-layer residual / fp32-output wiring:
  //   L0 transform : res=null, outF=R  (x1 becomes residual for block 0)
  //   L1 (1_0)     : res=null, outF=null
  //   L2 (2_0)     : res=R,    outF=R  (x2 becomes residual for block 1)
  //   L3 (1_1)     : res=null, outF=null
  //   L4 (2_1)     : res=R,    outF=Y  (final fp32 activations, in place)
  const float* resv[5] = {nullptr, nullptr, R, nullptr, R};
  float*       outv[5] = {R, nullptr, R, nullptr, Y};

  for (int L = 0; L < 5; ++L) {
    float* sum   = stat + L * 256 + 0;
    float* sq    = stat + L * 256 + 64;
    float* scale = stat + L * 256 + 128;
    float* shift = stat + L * 256 + 192;
    conv_wmma_kernel<<<256, 256, 0, stream>>>(A, Wb + (size_t)L * 4096, Y);
    stats_kernel<<<2048, 256, 0, stream>>>(Y, sum, sq);
    bnscale_kernel<<<1, 64, 0, stream>>>(sum, sq, g[L], be[L], scale, shift);
    bnapply_kernel<<<M_, 256, 0, stream>>>(Y, scale, shift, resv[L], outv[L], A);
  }

  maxpool_kernel<<<2048, 256, 0, stream>>>(Y, out);
}
